// MixedChunkAttentionLayer_23596550324709
// MI455X (gfx1250) — matrix operand order repaired and checked by reading, not compiled, not run on hardware
//
#include <hip/hip_runtime.h>
#include <hip/hip_bf16.h>
#include <math.h>

#define DEVI __device__ __forceinline__

typedef __bf16 bf16_t;
typedef __attribute__((ext_vector_type(16))) __bf16 v16bf;
typedef __attribute__((ext_vector_type(8)))  float  v8f;

namespace {
constexpr int Bsz = 8, Cch = 256, Tn = 8192, Gg = 128, QKd = 128, HIDd = 512;
constexpr int NG = Tn / Gg;   // 64 groups per batch
}

DEVI bf16_t tobf(float f) { return (bf16_t)f; }

DEVI unsigned int pk2(float a, float b) {
  union { bf16_t h[2]; unsigned int u; } x;
  x.h[0] = tobf(a); x.h[1] = tobf(b);
  return x.u;
}

DEVI float silu(float x) { return __fdividef(x, 1.f + __expf(-x)); }

DEVI v8f wmma_bf16(v16bf a, v16bf b, v8f c) {
  // D = A(16x32 bf16) x B(32x16 bf16) + C(16x16 f32)
  return __builtin_amdgcn_wmma_f32_16x16x32_bf16(false, a, false, b, (short)0, c,
                                                 false, false);
}

// ---- fragment loaders: all 16B-contiguous LDS reads ----------------------
// A 16x32 bf16, source row-major [m][k], ld multiple of 8, k0 multiple of 32.
// lane: m = m0 + (lane&15); K base += 8 for lanes 16-31.
// slots 0..7 -> k..k+7 (dwords 0..3), slots 8..15 -> k+16..k+23 (dwords 8..11)
DEVI v16bf frag_a_rm(const bf16_t* p, int ld, int m0, int k0, int lane) {
  const char* base = (const char*)(p + (size_t)(m0 + (lane & 15)) * ld + k0 +
                                   ((lane & 16) ? 8 : 0));
  uint4 lo = *(const uint4*)(base);
  uint4 hi = *(const uint4*)(base + 32);
  union { unsigned int u[8]; v16bf v; } x;
  x.u[0] = lo.x; x.u[1] = lo.y; x.u[2] = lo.z; x.u[3] = lo.w;
  x.u[4] = hi.x; x.u[5] = hi.y; x.u[6] = hi.z; x.u[7] = hi.w;
  return x.v;
}
// B 32x16 bf16 with B[k][n] = src[n*ld + k] (source stored [n][k]).
// lane: n = n0 + (lane&15); K base += 8 for lanes 16-31; slots 0..7 -> k..k+7, slots 8..15 -> k+16..k+23.
DEVI v16bf frag_b_tr(const bf16_t* p, int ld, int k0, int n0, int lane) {
  const char* base = (const char*)(p + (size_t)(n0 + (lane & 15)) * ld + k0 +
                                   ((lane & 16) ? 8 : 0));
  uint4 lo = *(const uint4*)(base);
  uint4 hi = *(const uint4*)(base + 32);
  union { unsigned int u[8]; v16bf v; } x;
  x.u[0] = lo.x; x.u[1] = lo.y; x.u[2] = lo.z; x.u[3] = lo.w;
  x.u[4] = hi.x; x.u[5] = hi.y; x.u[6] = hi.z; x.u[7] = hi.w;
  return x.v;
}

// generic global->LDS copy, dword granularity; colpairs = cols/2
DEVI void copy_tile2(bf16_t* dst, int dstld, const bf16_t* __restrict__ src,
                     int srcld, int rows, int colpairs) {
  int total = rows * colpairs;
  for (int idx = threadIdx.x; idx < total; idx += 256) {
    int r = idx / colpairs;
    int cp = (idx % colpairs) * 2;
    *reinterpret_cast<unsigned int*>(&dst[r * dstld + cp]) =
        *reinterpret_cast<const unsigned int*>(&src[(size_t)r * srcld + cp]);
  }
}

// ---- K0: batch all-zero mask fixup -> effective mask ---------------------
__global__ __launch_bounds__(256) void mask_fix_kernel(const float* __restrict__ masks,
                                                       float* __restrict__ effm) {
  int b = blockIdx.x;
  __shared__ float red[256];
  const float* mrow = masks + (size_t)b * Tn;
  float s = 0.f;
  for (int t = threadIdx.x; t < Tn; t += 256) s += mrow[t];
  red[threadIdx.x] = s;
  __syncthreads();
  for (int off = 128; off > 0; off >>= 1) {
    if ((int)threadIdx.x < off) red[threadIdx.x] += red[threadIdx.x + off];
    __syncthreads();
  }
  bool allz = (red[0] == 0.f);
  float* erow = effm + (size_t)b * Tn;
  for (int t = threadIdx.x; t < Tn; t += 256) erow[t] = allz ? 1.f : mrow[t];
}

// ---- K1: instance-norm stats over T per (b,c) ----------------------------
__global__ __launch_bounds__(256) void stats_kernel(const float* __restrict__ q,
                                                    float* __restrict__ mean,
                                                    float* __restrict__ rstd) {
  const int bc = blockIdx.x;
  const float* row = q + (size_t)bc * Tn;
  __shared__ float r1[256], r2[256];
  float s = 0.f, sq = 0.f;
  for (int t = threadIdx.x * 4; t < Tn; t += 1024) {
    float4 f = *(const float4*)(row + t);
    s += f.x + f.y + f.z + f.w;
    sq += f.x * f.x + f.y * f.y + f.z * f.z + f.w * f.w;
  }
  r1[threadIdx.x] = s; r2[threadIdx.x] = sq;
  __syncthreads();
  for (int off = 128; off > 0; off >>= 1) {
    if ((int)threadIdx.x < off) {
      r1[threadIdx.x] += r1[threadIdx.x + off];
      r2[threadIdx.x] += r2[threadIdx.x + off];
    }
    __syncthreads();
  }
  if (threadIdx.x == 0) {
    float m = r1[0] * (1.0f / Tn);
    float var = r2[0] * (1.0f / Tn) - m * m;
    mean[bc] = m;
    rstd[bc] = rsqrtf(var + 1e-5f);
  }
}

// ---- K2: weights f32 -> bf16, TRANSPOSED layouts -------------------------
// WgT/WvT: [HID][C], WqkT: [QK][C], WoT: [C][HID]
__global__ __launch_bounds__(256) void cvt_kernel(const float* __restrict__ Wg,
                                                  const float* __restrict__ Wv,
                                                  const float* __restrict__ Wqk,
                                                  const float* __restrict__ Wo,
                                                  bf16_t* gT, bf16_t* vT,
                                                  bf16_t* qkT, bf16_t* oT) {
  int i = blockIdx.x * 256 + threadIdx.x;  // grid 512 -> 131072 threads
  if (i < Cch * HIDd) {
    int n = i / Cch, k = i % Cch;                 // dest-major for W[C][HID]
    gT[i] = tobf(Wg[(size_t)k * HIDd + n]);
    vT[i] = tobf(Wv[(size_t)k * HIDd + n]);
    int c = i / HIDd, e = i % HIDd;               // WoT [C][HID] from Wo [HID][C]
    oT[i] = tobf(Wo[(size_t)e * Cch + c]);
  }
  if (i < Cch * QKd) {
    int n = i / Cch, k = i % Cch;
    qkT[i] = tobf(Wqk[(size_t)k * QKd + n]);
  }
}

// ---- K3: fused projection GEMM ------------------------------------------
// grid.x: 512 row tiles (128 tokens); grid.y: 0..3 gate, 4..7 vh, 8 qk heads
__global__ __launch_bounds__(256) void proj_kernel(
    const float* __restrict__ q, const float* __restrict__ v,
    const float* __restrict__ mean, const float* __restrict__ rstd,
    const bf16_t* __restrict__ WgT, const bf16_t* __restrict__ WvT,
    const bf16_t* __restrict__ WqkT,
    const float* __restrict__ bg, const float* __restrict__ bv,
    const float* __restrict__ bqk,
    const float* __restrict__ gamma, const float* __restrict__ beta,
    const float* __restrict__ effm,
    bf16_t* __restrict__ gateT, bf16_t* __restrict__ vhT,
    bf16_t* __restrict__ qqo, bf16_t* __restrict__ lqo,
    bf16_t* __restrict__ qko, bf16_t* __restrict__ lkT) {
  constexpr int LDA = 272, LDB = 264;
  __shared__ __align__(16) bf16_t As[128 * LDA];  // x tile [t][c]
  __shared__ __align__(16) bf16_t Bs[128 * LDB];  // W^T tile [n][c]

  const int ttile = blockIdx.x;
  const int y = blockIdx.y;
  const int b = ttile >> 6;
  const int t0 = (ttile & 63) << 7;
  const bool fromQ = (y < 4) || (y == 8);

  {  // A: channel pair per thread; transpose + inst-norm + bf16, dword stores
    const int u = threadIdx.x & 127;           // channels 2u, 2u+1
    const int th = (threadIdx.x >> 7) * 64;    // t half
    const int c0 = u * 2;
    const float* s0 = (fromQ ? q : v) + ((size_t)b * Cch + c0) * Tn + t0 + th;
    const float* s1 = s0 + Tn;
    float mu0 = 0.f, rs0 = 1.f, mu1 = 0.f, rs1 = 1.f;
    if (fromQ) {
      mu0 = mean[b * Cch + c0];     rs0 = rstd[b * Cch + c0];
      mu1 = mean[b * Cch + c0 + 1]; rs1 = rstd[b * Cch + c0 + 1];
    }
#pragma unroll 4
    for (int i = 0; i < 64; i += 4) {
      float4 f0 = *(const float4*)(s0 + i);
      float4 f1 = *(const float4*)(s1 + i);
      int r = th + i;
      *(unsigned int*)&As[(r + 0) * LDA + c0] = pk2((f0.x - mu0) * rs0, (f1.x - mu1) * rs1);
      *(unsigned int*)&As[(r + 1) * LDA + c0] = pk2((f0.y - mu0) * rs0, (f1.y - mu1) * rs1);
      *(unsigned int*)&As[(r + 2) * LDA + c0] = pk2((f0.z - mu0) * rs0, (f1.z - mu1) * rs1);
      *(unsigned int*)&As[(r + 3) * LDA + c0] = pk2((f0.w - mu0) * rs0, (f1.w - mu1) * rs1);
    }
  }
  const bf16_t* WT; int nbase;
  if (y < 4)      { WT = WgT;  nbase = y * 128; }
  else if (y < 8) { WT = WvT;  nbase = (y - 4) * 128; }
  else            { WT = WqkT; nbase = 0; }
  copy_tile2(Bs, LDB, WT + (size_t)nbase * Cch, Cch, 128, 128);
  __syncthreads();

  const int lane = threadIdx.x & 31;
  const int m0 = (threadIdx.x >> 5) * 16;
  v8f acc[8] = {};
  for (int kt = 0; kt < 256; kt += 32) {
    v16bf a = frag_a_rm(As, LDA, m0, kt, lane);
#pragma unroll
    for (int nt = 0; nt < 8; ++nt) {
      v16bf bm = frag_b_tr(Bs, LDB, kt, nt * 16, lane);
      acc[nt] = wmma_bf16(a, bm, acc[nt]);
    }
  }

  const int n_l = lane & 15;
  const int hi8 = (lane & 16) ? 8 : 0;
  if (y < 8) {  // gate^T / vh^T [b][e][t]: contiguous per-lane dword stores
    bf16_t* dstT = (y < 4) ? gateT : vhT;
    const float* bias = (y < 4) ? bg : bv;
#pragma unroll
    for (int nt = 0; nt < 8; ++nt) {
      int n = nbase + nt * 16 + n_l;
      float bb = bias[n];
      float s[8];
#pragma unroll
      for (int r = 0; r < 8; ++r) s[r] = silu(acc[nt][r] + bb);
      size_t base = ((size_t)b * HIDd + n) * Tn + t0 + m0 + hi8;
#pragma unroll
      for (int r = 0; r < 8; r += 2)
        *(unsigned int*)&dstT[base + r] = pk2(s[r], s[r + 1]);
    }
  } else {  // qk -> 4 OffsetScale heads; lin_k masked where mask != 0
#pragma unroll
    for (int nt = 0; nt < 8; ++nt) {
      int n = nt * 16 + n_l;
      float bb = bqk[n];
      float g0 = gamma[0 * QKd + n], g1 = gamma[1 * QKd + n];
      float g2 = gamma[2 * QKd + n], g3 = gamma[3 * QKd + n];
      float e0 = beta[0 * QKd + n], e1 = beta[1 * QKd + n];
      float e2 = beta[2 * QKd + n], e3 = beta[3 * QKd + n];
      float l3[8];
#pragma unroll
      for (int r = 0; r < 8; ++r) {
        int t = t0 + m0 + r + hi8;
        float s = silu(acc[nt][r] + bb);
        size_t o = ((size_t)b * Tn + t) * QKd + n;
        float kz = (effm[(size_t)b * Tn + t] == 0.f) ? 1.f : 0.f;
        qqo[o] = tobf(s * g0 + e0);
        lqo[o] = tobf(s * g1 + e1);
        qko[o] = tobf(s * g2 + e2);
        l3[r] = (s * g3 + e3) * kz;
      }
      size_t base = ((size_t)b * QKd + n) * Tn + t0 + m0 + hi8;  // lk^T [b][d][t]
#pragma unroll
      for (int r = 0; r < 8; r += 2)
        *(unsigned int*)&lkT[base + r] = pk2(l3[r], l3[r + 1]);
    }
  }
}

// ---- K4: per-(b, group, e-tile) quadratic + linear attention -------------
__global__ __launch_bounds__(256) void attn_kernel(
    const bf16_t* __restrict__ qq, const bf16_t* __restrict__ qkk,
    const bf16_t* __restrict__ lq, const bf16_t* __restrict__ lkT,
    const bf16_t* __restrict__ vhT, const bf16_t* __restrict__ gateT,
    const float* __restrict__ effm, bf16_t* __restrict__ hout) {
  constexpr int LD = 136;
  __shared__ __align__(16) bf16_t sb0[128 * LD];
  __shared__ __align__(16) bf16_t sb1[128 * LD];
  __shared__ __align__(16) bf16_t sb2[128 * LD];
  __shared__ __align__(16) bf16_t sb3[128 * LD];

  const int e0 = blockIdx.x << 7;
  const int g = blockIdx.y;
  const int b = blockIdx.z;
  const size_t tbase = (size_t)b * Tn + (size_t)g * Gg;

  const int lane = threadIdx.x & 31;
  const int m0 = (threadIdx.x >> 5) * 16;
  const int n_l = lane & 15;
  const int hi8 = (lane & 16) ? 8 : 0;

  // prefetch tiles needed by later phases (global_prefetch_b8)
  __builtin_prefetch(qq + tbase * QKd + (size_t)threadIdx.x * 64, 0, 1);
  __builtin_prefetch(qkk + tbase * QKd + (size_t)threadIdx.x * 64, 0, 1);
  __builtin_prefetch(lq + tbase * QKd + (size_t)threadIdx.x * 64, 0, 1);

  // sb0 = lk^T [d][j], sb1 = vg^T [e][j]
  copy_tile2(sb0, LD, lkT + (size_t)b * QKd * Tn + (size_t)g * Gg, Tn, 128, 64);
  copy_tile2(sb1, LD, vhT + ((size_t)b * HIDd + e0) * Tn + (size_t)g * Gg, Tn, 128, 64);
  __syncthreads();

  // lin_kv[d][e] = sum_j lk[j][d] * vg[j][e] / T
  v8f kvacc[8] = {};
  for (int kt = 0; kt < 128; kt += 32) {
    v16bf a = frag_a_rm(sb0, LD, m0, kt, lane);       // A[d][j]
#pragma unroll
    for (int nt = 0; nt < 8; ++nt) {
      v16bf bm = frag_b_tr(sb1, LD, kt, nt * 16, lane);  // B[j][e] from [e][j]
      kvacc[nt] = wmma_bf16(a, bm, kvacc[nt]);
    }
  }
  __syncthreads();
  // store lin_kv transposed [e][d]: per-lane d-contiguous dword stores
#pragma unroll
  for (int nt = 0; nt < 8; ++nt) {
    int base = (nt * 16 + n_l) * LD + m0 + hi8;
#pragma unroll
    for (int r = 0; r < 8; r += 2)
      *(unsigned int*)&sb2[base + r] =
          pk2(kvacc[nt][r] * (1.0f / Tn), kvacc[nt][r + 1] * (1.0f / Tn));
  }
  copy_tile2(sb3, LD, qq + tbase * QKd, QKd, 128, 64);   // qq  [i][d]
  copy_tile2(sb0, LD, qkk + tbase * QKd, QKd, 128, 64);  // qk  [j][d]
  __syncthreads();

  // sim[i][j] = qq . qk^T
  v8f sacc[8] = {};
  for (int kt = 0; kt < 128; kt += 32) {
    v16bf a = frag_a_rm(sb3, LD, m0, kt, lane);
#pragma unroll
    for (int nt = 0; nt < 8; ++nt) {
      v16bf bm = frag_b_tr(sb0, LD, kt, nt * 16, lane);  // B[d][j] from [j][d]
      sacc[nt] = wmma_bf16(a, bm, sacc[nt]);
    }
  }
  __syncthreads();

  // laplace attention + column mask (keep where mask == 0) -> attn in sb0
  const float MU = 0.70710678118654752f;    // sqrt(0.5)
  const float LINV = 0.79788456080286536f;  // 1/(std*sqrt(2)) = 1/sqrt(0.5*pi)
#pragma unroll
  for (int nt = 0; nt < 8; ++nt) {
    int j = nt * 16 + n_l;
    float cm = (effm[tbase + j] == 0.f) ? 1.f : 0.f;
#pragma unroll
    for (int r = 0; r < 8; ++r) {
      float s = sacc[nt][r] * (1.0f / Gg);
      float at = 0.5f * (1.f + erff((s - MU) * LINV)) * cm;
      sb0[(m0 + r + hi8) * LD + j] = tobf(at);
    }
  }
  copy_tile2(sb3, LD, lq + tbase * QKd, QKd, 128, 64);  // lq [i][d]
  __syncthreads();

  // acc = attn @ vg + lq @ lin_kv
  v8f facc[8] = {};
  for (int kt = 0; kt < 128; kt += 32) {
    v16bf aat = frag_a_rm(sb0, LD, m0, kt, lane);
    v16bf alq = frag_a_rm(sb3, LD, m0, kt, lane);
#pragma unroll
    for (int nt = 0; nt < 8; ++nt) {
      v16bf bv1 = frag_b_tr(sb1, LD, kt, nt * 16, lane);  // vg
      facc[nt] = wmma_bf16(aat, bv1, facc[nt]);
      v16bf bkv = frag_b_tr(sb2, LD, kt, nt * 16, lane);  // lin_kv
      facc[nt] = wmma_bf16(alq, bkv, facc[nt]);
    }
  }
  // h = gate * (quad_out + lin_out); gate^T read is contiguous per lane
#pragma unroll
  for (int nt = 0; nt < 8; ++nt) {
    int e = e0 + nt * 16 + n_l;
    size_t gbase = ((size_t)b * HIDd + e) * Tn + (size_t)g * Gg + m0 + hi8;
    union { uint4 u; bf16_t h[8]; } gx;
    gx.u = *(const uint4*)(gateT + gbase);
#pragma unroll
    for (int r = 0; r < 8; ++r) {
      int m = m0 + r + hi8;
      size_t o = (tbase + m) * HIDd + e;
      hout[o] = tobf(facc[nt][r] * (float)gx.h[r]);
    }
  }
}

// ---- K5: out[c][t] = (h @ Wo + bo)^T * mask ------------------------------
// D[m=c][n=t]: A = Wo^T [c][e] (frag_a_rm), B[k=e][n=t] = h[t][e] (frag_b_tr)
__global__ __launch_bounds__(256) void out_kernel(const bf16_t* __restrict__ h,
                                                  const bf16_t* __restrict__ WoT,
                                                  const float* __restrict__ bo,
                                                  const float* __restrict__ effm,
                                                  float* __restrict__ out) {
  constexpr int LD = 136;
  __shared__ __align__(16) char smem[69632];
  bf16_t* As = reinterpret_cast<bf16_t*>(smem);
  bf16_t* Bs = reinterpret_cast<bf16_t*>(smem + 128 * LD * 2);
  float* Cs = reinterpret_cast<float*>(smem);  // reused after final barrier

  const int ttile = blockIdx.x;
  const int b = ttile >> 6;
  const int t0 = (ttile & 63) << 7;
  const int c0 = blockIdx.y << 7;

  const int lane = threadIdx.x & 31;
  const int m0 = (threadIdx.x >> 5) * 16;
  const int n_l = lane & 15;
  const int hi8 = (lane & 16) ? 8 : 0;

  v8f acc[8] = {};
  for (int kb = 0; kb < HIDd; kb += 128) {
    if (kb + 128 < HIDd) {  // prefetch next h k-tile
      int rr = threadIdx.x >> 1;
      __builtin_prefetch(h + ((size_t)b * Tn + t0 + rr) * HIDd + kb + 128 +
                             (threadIdx.x & 1) * 64, 0, 1);
    }
    copy_tile2(As, LD, WoT + (size_t)c0 * HIDd + kb, HIDd, 128, 64);
    copy_tile2(Bs, LD, h + ((size_t)b * Tn + t0) * HIDd + kb, HIDd, 128, 64);
    __syncthreads();
    for (int kt = 0; kt < 128; kt += 32) {
      v16bf a = frag_a_rm(As, LD, m0, kt, lane);
#pragma unroll
      for (int nt = 0; nt < 8; ++nt) {
        v16bf bm = frag_b_tr(Bs, LD, kt, nt * 16, lane);
        acc[nt] = wmma_bf16(a, bm, acc[nt]);
      }
    }
    __syncthreads();
  }
  // epilogue: bias + mask, stage [c][t] in LDS (m=c, n=t)
  float em[8], bor[8];
#pragma unroll
  for (int nt = 0; nt < 8; ++nt) em[nt] = effm[(size_t)b * Tn + t0 + nt * 16 + n_l];
#pragma unroll
  for (int r = 0; r < 8; ++r) bor[r] = bo[c0 + m0 + hi8 + r];
#pragma unroll
  for (int nt = 0; nt < 8; ++nt) {
    int tl = nt * 16 + n_l;
#pragma unroll
    for (int r = 0; r < 8; ++r)
      Cs[(m0 + hi8 + r) * 132 + tl] = (acc[nt][r] + bor[r]) * em[nt];
  }
  __syncthreads();
  {  // contiguous stores along t per channel row
    int rw = threadIdx.x >> 1;
    int seg = (threadIdx.x & 1) << 6;
    float* op = out + ((size_t)b * Cch + c0 + rw) * Tn + t0 + seg;
    const float* cp = Cs + rw * 132 + seg;
#pragma unroll
    for (int i = 0; i < 64; i += 4)
      *reinterpret_cast<float4*>(op + i) = *reinterpret_cast<const float4*>(cp + i);
  }
}

extern "C" void kernel_launch(void* const* d_in, const int* in_sizes, int n_in,
                              void* d_out, int out_size, void* d_ws, size_t ws_size,
                              hipStream_t stream) {
  const float* q = (const float*)d_in[0];
  // d_in[1] (k) unused by the reference
  const float* v = (const float*)d_in[2];
  const float* masks = (const float*)d_in[3];
  const float* Wg = (const float*)d_in[4];
  const float* bg = (const float*)d_in[5];
  const float* Wv = (const float*)d_in[6];
  const float* bv = (const float*)d_in[7];
  const float* Wqk = (const float*)d_in[8];
  const float* bqk = (const float*)d_in[9];
  const float* gamma = (const float*)d_in[10];
  const float* beta = (const float*)d_in[11];
  const float* Wo = (const float*)d_in[12];
  const float* bo = (const float*)d_in[13];
  float* out = (float*)d_out;

  char* w = (char*)d_ws;
  auto carve = [&](size_t bytes) -> char* {
    char* p = w;
    w += (bytes + 255) & ~(size_t)255;
    return p;
  };
  float* effm = (float*)carve((size_t)Bsz * Tn * 4);
  float* mean = (float*)carve((size_t)Bsz * Cch * 4);
  float* rstd = (float*)carve((size_t)Bsz * Cch * 4);
  bf16_t* WgT = (bf16_t*)carve((size_t)Cch * HIDd * 2);
  bf16_t* WvT = (bf16_t*)carve((size_t)Cch * HIDd * 2);
  bf16_t* WqkT = (bf16_t*)carve((size_t)Cch * QKd * 2);
  bf16_t* WoT = (bf16_t*)carve((size_t)HIDd * Cch * 2);
  bf16_t* gateT = (bf16_t*)carve((size_t)Bsz * Tn * HIDd * 2);
  bf16_t* vhT = (bf16_t*)carve((size_t)Bsz * Tn * HIDd * 2);
  bf16_t* qqb = (bf16_t*)carve((size_t)Bsz * Tn * QKd * 2);
  bf16_t* lqb = (bf16_t*)carve((size_t)Bsz * Tn * QKd * 2);
  bf16_t* qkb = (bf16_t*)carve((size_t)Bsz * Tn * QKd * 2);
  bf16_t* lkT = (bf16_t*)carve((size_t)Bsz * Tn * QKd * 2);
  bf16_t* hb = (bf16_t*)carve((size_t)Bsz * Tn * HIDd * 2);

  mask_fix_kernel<<<Bsz, 256, 0, stream>>>(masks, effm);
  stats_kernel<<<Bsz * Cch, 256, 0, stream>>>(q, mean, rstd);
  cvt_kernel<<<512, 256, 0, stream>>>(Wg, Wv, Wqk, Wo, WgT, WvT, WqkT, WoT);
  proj_kernel<<<dim3(Bsz * Tn / 128, 9), 256, 0, stream>>>(
      q, v, mean, rstd, WgT, WvT, WqkT, bg, bv, bqk, gamma, beta, effm, gateT,
      vhT, qqb, lqb, qkb, lkT);
  attn_kernel<<<dim3(HIDd / 128, NG, Bsz), 256, 0, stream>>>(qqb, qkb, lqb, lkT,
                                                             vhT, gateT, effm, hb);
  out_kernel<<<dim3(Bsz * Tn / 128, Cch / 128), 256, 0, stream>>>(hb, WoT, bo,
                                                                  effm, out);
}
